// EnetGnn_69810398429294
// MI455X (gfx1250) — matrix operand order repaired and checked by reading, not compiled, not run on hardware
//
#include <hip/hip_runtime.h>
#include <hip/hip_bf16.h>

// ---------------------------------------------------------------------------
// Types for CDNA5 WMMA (wave32): v_wmma_f32_16x16x32_bf16
// ---------------------------------------------------------------------------
typedef __attribute__((ext_vector_type(16))) __bf16          v16bf;
typedef __attribute__((ext_vector_type(8)))  float           v8f;
typedef __attribute__((ext_vector_type(16))) unsigned short  v16us;
typedef __attribute__((ext_vector_type(8)))  unsigned short  v8us;

static __device__ __forceinline__ unsigned short f2bf(float f) {
    unsigned u = __float_as_uint(f);
    unsigned r = u + 0x7FFFu + ((u >> 16) & 1u);   // round-to-nearest-even
    return (unsigned short)(r >> 16);
}
static __device__ __forceinline__ float bf2f(unsigned short b) {
    return __uint_as_float(((unsigned)b) << 16);
}
static __device__ __forceinline__ unsigned short bfsub(unsigned short a, unsigned short b) {
    return f2bf(bf2f(a) - bf2f(b));
}
static __device__ __forceinline__ v16us cmb(v8us lo, v8us hi) {
    v16us r;
#pragma unroll
    for (int i = 0; i < 8; ++i) { r[i] = lo[i]; r[i + 8] = hi[i]; }
    return r;
}
static __device__ __forceinline__ v8f wmma_bf16(v16us a, v16us b, v8f c) {
    return __builtin_amdgcn_wmma_f32_16x16x32_bf16(
        false, __builtin_bit_cast(v16bf, a),
        false, __builtin_bit_cast(v16bf, b),
        (short)0, c, false, false);
}

// ---------------------------------------------------------------------------
// Constants: N=4, C=128, pooled H=W=64, HW=4096, k=16
// ---------------------------------------------------------------------------
#define HW     4096
#define CCH    128
#define KNN    16

// ---------------------------------------------------------------------------
// K0: zero a float buffer
// ---------------------------------------------------------------------------
__global__ void zero_kernel(float* __restrict__ p, int n) {
    int i = blockIdx.x * 256 + threadIdx.x;
    if (i < n) p[i] = 0.0f;
}

// ---------------------------------------------------------------------------
// K1: fused 2x2 maxpool + transpose to (N,HW,C) f32 + L2-normalized bf16 copy
// grid (256, 2): block = (n,i) output row strip, 256 threads
// ---------------------------------------------------------------------------
__global__ __launch_bounds__(256) void pool_norm_kernel(
    const float* __restrict__ in_rgb, const float* __restrict__ in_ir,
    float* __restrict__ x_rgb, float* __restrict__ x_ir,
    unsigned short* __restrict__ xn_rgb, unsigned short* __restrict__ xn_ir)
{
    __shared__ float xt[128 * 65];
    __shared__ float invn[64];
    const float* in = blockIdx.y ? in_ir : in_rgb;
    float* xo = blockIdx.y ? x_ir : x_rgb;
    unsigned short* xn = blockIdx.y ? xn_ir : xn_rgb;

    const int n = blockIdx.x >> 6;
    const int i = blockIdx.x & 63;
    const int t = threadIdx.x;

    for (int cc = 0; cc < 16; ++cc) {
#pragma unroll
        for (int s = 0; s < 2; ++s) {
            int o = t + 256 * s;            // 0..511
            int cl = o >> 6, j = o & 63;
            int c = cc * 8 + cl;
            const float* p = in + ((size_t)(n * 128 + c) * 128 + 2 * i) * 128 + 2 * j;
            float2 a = *(const float2*)p;
            float2 b = *(const float2*)(p + 128);
            xt[c * 65 + j] = fmaxf(fmaxf(a.x, a.y), fmaxf(b.x, b.y));
        }
    }
    __syncthreads();
    if (t < 64) {
        float ss = 0.0f;
        for (int c = 0; c < 128; ++c) { float v = xt[c * 65 + t]; ss += v * v; }
        invn[t] = 1.0f / fmaxf(sqrtf(ss), 1e-12f);
    }
    __syncthreads();
    const size_t base = ((size_t)n * HW + (size_t)i * 64) * CCH;
    for (int s = 0; s < 32; ++s) {
        int o = t + 256 * s;                // 0..8191
        int c = o & 127, j = o >> 7;
        float v = xt[c * 65 + j];
        xo[base + (size_t)j * CCH + c] = v;
        xn[base + (size_t)j * CCH + c] = f2bf(v * invn[j]);
    }
}

// ---------------------------------------------------------------------------
// K2: swizzle W (256x128, row-major (in,out)) into per-lane contiguous WMMA
// B-fragments: wz[((kb*8+nt)*32+lane)*16 + j] = bf16( W[(kb*32+hi*8+j+(j>=8?8:0))*128 + n] )
// grid (8, 2), 256 threads
// ---------------------------------------------------------------------------
__global__ void wswz_kernel(const float* __restrict__ Wr, const float* __restrict__ Wi,
                            unsigned short* __restrict__ wz)
{
    int t = blockIdx.x * 256 + threadIdx.x;    // 0..2047
    const float* W = blockIdx.y ? Wi : Wr;
    unsigned short* o = wz + (size_t)blockIdx.y * 32768 + (size_t)t * 16;
    int lane = t & 31, nt = (t >> 5) & 7, kb = t >> 8;
    int n = nt * 16 + (lane & 15), hi = lane >> 4;
#pragma unroll
    for (int j = 0; j < 16; ++j)
        o[j] = f2bf(W[(kb * 32 + hi * 8 + j + (j >= 8 ? 8 : 0)) * 128 + n]);
}

// ---------------------------------------------------------------------------
// K3: KNN top-16 via WMMA gram sweep.
// grid (128, 2); 256 threads = 8 waves; wave = one 16-query strip, sweeps all
// 4096 candidates of its batch with ping-pong double-buffered B fragments.
// Selection uses packed 32-bit keys: ordered-float sim bits [31:12] | cand [11:0]
// so the sorted top-16 insert is a pure v_max_u32/v_min_u32 bubble (no cndmask
// chains, no register-array copies).
// ---------------------------------------------------------------------------
static __device__ __forceinline__ unsigned simkey(float f, int idx) {
    unsigned u = __float_as_uint(f);
    u ^= (unsigned)((int)u >> 31) | 0x80000000u;      // monotone float->uint
    return (u & 0xFFFFF000u) | (unsigned)idx;          // idx in [0,4096)
}
static __device__ __forceinline__ void key_insert(unsigned (&bk)[16], unsigned k) {
    if (k <= bk[15]) return;                           // below current 16th best
#pragma unroll
    for (int s = 0; s < 16; ++s) {
        unsigned old = bk[s];
        bk[s] = (old > k) ? old : k;                   // v_max_u32
        k     = (old > k) ? k : old;                   // v_min_u32
    }
}

__global__ __launch_bounds__(256) void knn_kernel(
    const unsigned short* __restrict__ xn_rgb, const unsigned short* __restrict__ xn_ir,
    int* __restrict__ knn_rgb, int* __restrict__ knn_ir)
{
    const unsigned short* xn = blockIdx.y ? xn_ir : xn_rgb;
    int* knn = blockIdx.y ? knn_ir : knn_rgb;

    __shared__ float    sim[8][256];
    __shared__ unsigned lkey[8][32][16];

    const int wave = threadIdx.x >> 5;
    const int lane = threadIdx.x & 31;
    const int m = lane & 15;
    const int hi = lane >> 4;
    const int q0 = blockIdx.x * 128 + wave * 16;     // global query row base
    const int cbase = (q0 >> 12) << 12;              // batch candidate base

    // A fragments (16 queries x 32 K) for all 4 K-blocks, held in registers.
    const unsigned short* arow = xn + (size_t)(q0 + m) * CCH;
    v16us af[4];
#pragma unroll
    for (int kb = 0; kb < 4; ++kb) {
        v8us lo = *(const v8us*)(arow + kb * 32 + hi * 8);
        v8us hp = *(const v8us*)(arow + kb * 32 + 16 + hi * 8);
        af[kb] = cmb(lo, hp);
    }

    unsigned bk[16];
#pragma unroll
    for (int s = 0; s < 16; ++s) bk[s] = 0u;

    v8us cb[8], nb[8];                                // ping-pong B buffers

    auto loadB = [&](int jt, v8us (&B)[8]) {
        const unsigned short* brow = xn + (size_t)(cbase + jt * 16 + m) * CCH;
#pragma unroll
        for (int kb = 0; kb < 4; ++kb) {
            B[kb * 2]     = *(const v8us*)(brow + kb * 32 + hi * 8);
            B[kb * 2 + 1] = *(const v8us*)(brow + kb * 32 + 16 + hi * 8);
        }
    };
    auto computeTile = [&](int jt, v8us (&B)[8]) {
        v8f acc;
#pragma unroll
        for (int e = 0; e < 8; ++e) acc[e] = 0.0f;
#pragma unroll
        for (int kb = 0; kb < 4; ++kb)
            acc = wmma_bf16(af[kb], cmb(B[kb * 2], B[kb * 2 + 1]), acc);
        // D layout: lane holds column N=m, rows M = v + 8*hi
#pragma unroll
        for (int v = 0; v < 8; ++v) sim[wave][(v + hi * 8) * 16 + m] = acc[v];
        __builtin_amdgcn_wave_barrier();
        const int c0 = hi * 8;   // lane scans query row m, candidate half c0..c0+7
#pragma unroll
        for (int c = 0; c < 8; ++c) {
            float v = sim[wave][m * 16 + c0 + c];
            key_insert(bk, simkey(v, jt * 16 + c0 + c));
        }
        __builtin_amdgcn_wave_barrier();
    };

    loadB(0, cb);
    for (int jt = 0; jt < 256; jt += 2) {
        loadB(jt + 1, nb);                 // in flight while cb computes
        computeTile(jt, cb);
        loadB((jt + 2) & 255, cb);         // wraps harmlessly on last iter
        computeTile(jt + 1, nb);
    }

#pragma unroll
    for (int s = 0; s < 16; ++s) lkey[wave][lane][s] = bk[s];
    __builtin_amdgcn_wave_barrier();

    if (lane < 16) {   // merge the two sorted key lists for query row `lane`
        int pa = 0, pb = 0;
        int* o = knn + (size_t)(q0 + lane) * KNN;
#pragma unroll
        for (int s = 0; s < 16; ++s) {
            unsigned ka = lkey[wave][lane][pa];
            unsigned kb2 = lkey[wave][lane + 16][pb];
            if (ka >= kb2) { o[s] = (int)(ka & 0xFFFu); ++pa; }
            else           { o[s] = (int)(kb2 & 0xFFFu); ++pb; }
        }
    }
}

// ---------------------------------------------------------------------------
// K4: GNN GEMM + mean reduction.
// grid 4096, 128 threads = 4 waves. Block handles 64 (row,k) pairs (= 4 query
// rows). Stages gathered batch-0 neighbor rows (both modalities) in LDS as
// bf16, builds the concat/diff A fragments on the fly, accumulates
// leaky(A@W+b) column sums into per-batch accumulators.
// ---------------------------------------------------------------------------
#define NBSTRIDE 136   // 128 + 8 pad: 16B aligned rows, conflict-free b128 reads

__global__ __launch_bounds__(128) void gnn_kernel(
    const float* __restrict__ x_rgb, const float* __restrict__ x_ir,
    const int* __restrict__ knn_rgb, const int* __restrict__ knn_ir,
    const unsigned short* __restrict__ wz,
    const float* __restrict__ b_rgb, const float* __restrict__ b_ir,
    float* __restrict__ sumf)
{
    __shared__ unsigned short NR[64 * NBSTRIDE];
    __shared__ unsigned short NI[64 * NBSTRIDE];
    __shared__ float lsum[256];

    const int t = threadIdx.x;
    lsum[t] = 0.0f; lsum[t + 128] = 0.0f;

    const int p0 = blockIdx.x * 64;   // base (row,k) pair index
    {
        const int j = t >> 1, h = t & 1;
        const int p = p0 + j;
        const float* vr = x_rgb + (size_t)knn_rgb[p] * CCH + h * 64;  // batch-0 gather
        const float* vi = x_ir  + (size_t)knn_ir[p]  * CCH + h * 64;
        unsigned short* nr = NR + j * NBSTRIDE + h * 64;
        unsigned short* ni = NI + j * NBSTRIDE + h * 64;
#pragma unroll
        for (int c4 = 0; c4 < 16; ++c4) {
            float4 a = *(const float4*)(vr + c4 * 4);
            float4 b = *(const float4*)(vi + c4 * 4);
            nr[c4 * 4 + 0] = f2bf(a.x); nr[c4 * 4 + 1] = f2bf(a.y);
            nr[c4 * 4 + 2] = f2bf(a.z); nr[c4 * 4 + 3] = f2bf(a.w);
            ni[c4 * 4 + 0] = f2bf(b.x); ni[c4 * 4 + 1] = f2bf(b.y);
            ni[c4 * 4 + 2] = f2bf(b.z); ni[c4 * 4 + 3] = f2bf(b.w);
        }
    }
    __syncthreads();

    const int wave = t >> 5, lane = t & 31, m = lane & 15, hi = lane >> 4;
    const int roff = (wave * 16 + m) * NBSTRIDE;

#pragma unroll 1
    for (int mod = 0; mod < 2; ++mod) {
        const unsigned short* S = mod ? NI : NR;   // self
        const unsigned short* O = mod ? NR : NI;   // other
        const unsigned short* Wm = wz + (size_t)mod * 32768;
        const float* bias = mod ? b_ir : b_rgb;

        v8f acc[8];
#pragma unroll
        for (int nt = 0; nt < 8; ++nt)
#pragma unroll
            for (int e = 0; e < 8; ++e) acc[nt][e] = 0.0f;

#pragma unroll
        for (int kb = 0; kb < 8; ++kb) {
            v16us afr;
            if (kb < 4) {                       // A cols 0..127 : nb_self
                const unsigned short* p = S + roff + kb * 32;
                afr = cmb(*(const v8us*)(p + hi * 8), *(const v8us*)(p + 16 + hi * 8));
            } else {                            // A cols 128..255 : nb_self - nb_other
                const int kk = (kb - 4) * 32;
                const unsigned short* ps = S + roff + kk;
                const unsigned short* po = O + roff + kk;
                v8us s0 = *(const v8us*)(ps + hi * 8), s1 = *(const v8us*)(ps + 16 + hi * 8);
                v8us o0 = *(const v8us*)(po + hi * 8), o1 = *(const v8us*)(po + 16 + hi * 8);
#pragma unroll
                for (int e = 0; e < 8; ++e) {
                    afr[e]     = bfsub(s0[e], o0[e]);
                    afr[e + 8] = bfsub(s1[e], o1[e]);
                }
            }
#pragma unroll
            for (int nt = 0; nt < 8; ++nt) {
                const v8us* bp = (const v8us*)(Wm + ((size_t)(kb * 8 + nt) * 32 + lane) * 16);
                acc[nt] = wmma_bf16(afr, cmb(bp[0], bp[1]), acc[nt]);
            }
        }
        // bias + leaky, column-reduce over the 16 neighbor rows of this wave
#pragma unroll
        for (int nt = 0; nt < 8; ++nt) {
            const int n = nt * 16 + m;
            const float bb = bias[n];
            float s = 0.0f;
#pragma unroll
            for (int v = 0; v < 8; ++v) {
                float f = acc[nt][v] + bb;
                s += (f >= 0.0f) ? f : 0.01f * f;
            }
            s += __shfl_xor(s, 16, 32);
            if (hi == 0) atomicAdd(&lsum[mod * 128 + n], s);
        }
    }
    __syncthreads();
    const int batch = p0 >> 16;     // 65536 pairs per batch
    atomicAdd(&sumf[batch * 256 + t], lsum[t]);
    atomicAdd(&sumf[batch * 256 + 128 + t], lsum[t + 128]);
}

// ---------------------------------------------------------------------------
// K5: SE gate per batch. grid 4, 128 threads.
// ---------------------------------------------------------------------------
__global__ void se_kernel(const float* __restrict__ sumf,
                          const float* __restrict__ w1, const float* __restrict__ b1,
                          const float* __restrict__ w2, const float* __restrict__ b2,
                          float* __restrict__ gate)
{
    __shared__ float mm[256];
    __shared__ float z[8];
    const int nb = blockIdx.x, t = threadIdx.x;
    const float scale = 1.0f / ((float)HW * (float)KNN);
    mm[t]       = sumf[nb * 256 + t] * scale;
    mm[t + 128] = sumf[nb * 256 + 128 + t] * scale;
    __syncthreads();
    if (t < 8) {
        float s = b1[t];
        for (int i = 0; i < 256; ++i) s += mm[i] * w1[i * 8 + t];
        z[t] = (s >= 0.0f) ? s : 0.01f * s;
    }
    __syncthreads();
    float s = b2[t];
#pragma unroll
    for (int j = 0; j < 8; ++j) s += z[j] * w2[j * 128 + t];
    gate[nb * 128 + t] = 1.0f / (1.0f + expf(-s));
}

// ---------------------------------------------------------------------------
// K6: blend + ReLU, write NCHW output. grid 8192, 256 threads.
// ---------------------------------------------------------------------------
__global__ void out_kernel(const float* __restrict__ x_rgb, const float* __restrict__ x_ir,
                           const float* __restrict__ gate,
                           const float* __restrict__ g1p, const float* __restrict__ g2p,
                           float* __restrict__ out)
{
    size_t o = (size_t)blockIdx.x * 256 + threadIdx.x;   // over N*C*HW = 2^21
    int hw = (int)(o & 4095);
    int c  = (int)((o >> 12) & 127);
    int n  = (int)(o >> 19);
    float vr = x_rgb[(((size_t)n * HW + hw) << 7) + c];
    float vi = x_ir [(((size_t)n * HW + hw) << 7) + c];
    float g = gate[n * 128 + c];
    float h = g1p[0] * g * vr + g2p[0] * (1.0f - g) * vi;
    out[o] = fmaxf(h, 0.0f);
}

// ---------------------------------------------------------------------------
// Launch
// ---------------------------------------------------------------------------
extern "C" void kernel_launch(void* const* d_in, const int* in_sizes, int n_in,
                              void* d_out, int out_size, void* d_ws, size_t ws_size,
                              hipStream_t stream)
{
    (void)in_sizes; (void)n_in; (void)out_size; (void)ws_size;

    const float* rgb = (const float*)d_in[0];
    const float* ir  = (const float*)d_in[1];
    const float* Wrg = (const float*)d_in[2];
    const float* brg = (const float*)d_in[3];
    const float* Wig = (const float*)d_in[4];
    const float* big = (const float*)d_in[5];
    const float* sw1 = (const float*)d_in[6];
    const float* sb1 = (const float*)d_in[7];
    const float* sw2 = (const float*)d_in[8];
    const float* sb2 = (const float*)d_in[9];
    const float* g1  = (const float*)d_in[10];
    const float* g2  = (const float*)d_in[11];
    float* out = (float*)d_out;

    char* ws = (char*)d_ws;
    float*          x_rgb  = (float*)(ws);                         //  8 MB
    float*          x_ir   = (float*)(ws + 8388608);               //  8 MB
    unsigned short* xn_rgb = (unsigned short*)(ws + 16777216);     //  4 MB
    unsigned short* xn_ir  = (unsigned short*)(ws + 20971520);     //  4 MB
    int*            knn_r  = (int*)(ws + 25165824);                //  1 MB
    int*            knn_i  = (int*)(ws + 26214400);                //  1 MB
    unsigned short* wz     = (unsigned short*)(ws + 27262976);     // 128 KB
    float*          sumf   = (float*)(ws + 27328512);              //  4 KB
    float*          gate   = (float*)(ws + 27332608);              //  2 KB

    zero_kernel<<<4, 256, 0, stream>>>(sumf, 1024);
    pool_norm_kernel<<<dim3(256, 2), 256, 0, stream>>>(rgb, ir, x_rgb, x_ir, xn_rgb, xn_ir);
    wswz_kernel<<<dim3(8, 2), 256, 0, stream>>>(Wrg, Wig, wz);
    knn_kernel<<<dim3(128, 2), 256, 0, stream>>>(xn_rgb, xn_ir, knn_r, knn_i);
    gnn_kernel<<<4096, 128, 0, stream>>>(x_rgb, x_ir, knn_r, knn_i, wz, brg, big, sumf);
    se_kernel<<<4, 128, 0, stream>>>(sumf, sw1, sb1, sw2, sb2, gate);
    out_kernel<<<8192, 256, 0, stream>>>(x_rgb, x_ir, gate, g1, g2, out);
}
